// Attention_67336497266819
// MI455X (gfx1250) — matrix operand order repaired and checked by reading, not compiled, not run on hardware
//
#include <hip/hip_runtime.h>
#include <hip/hip_bf16.h>

typedef __attribute__((ext_vector_type(16))) _Float16 v16h;
typedef __attribute__((ext_vector_type(8)))  float    v8f;

#define EPSF 1e-5f

// ---------------------------------------------------------------------------
// WMMA fragment loaders (v_wmma_f32_16x16x32_f16 layouts, CDNA5 ISA 7.12.2)
// A (16Mx32K, row-major src, ld in elements):
//   lane l (0-15): row M=l, K = {0..7, 16..23}; lane l+16: row M=l, K = {8..15, 24..31}
// B (32Kx16N) loaded from Bt storage where Bt[n][k] (i.e. weights stored [N][K]):
//   lane l (0-15): col N=l, K = {0..7, 16..23}; lane l+16: col N=l, K = {8..15, 24..31}
// ---------------------------------------------------------------------------
__device__ __forceinline__ v16h ld_a16x32(const _Float16* __restrict__ p, int ld, int lane) {
  const int l16 = lane & 15, hl = lane >> 4;
  const _Float16* r = p + l16 * ld + hl * 8;
  v16h out;
  *((uint4*)&out)     = *(const uint4*)(r);
  *((uint4*)&out + 1) = *(const uint4*)(r + 16);
  return out;
}

__device__ __forceinline__ v16h ld_b32x16(const _Float16* __restrict__ p, int ld, int lane) {
  const int l16 = lane & 15, hl = lane >> 4;
  const _Float16* r = p + l16 * ld + hl * 8;
  v16h out;
  *((uint4*)&out)     = *(const uint4*)(r);
  *((uint4*)&out + 1) = *(const uint4*)(r + 16);
  return out;
}

// ---------------------------------------------------------------------------
// Elementwise prep kernels
// ---------------------------------------------------------------------------
__global__ void k_cvt_f16(const float* __restrict__ src, _Float16* __restrict__ dst, int n) {
  int i = blockIdx.x * blockDim.x + threadIdx.x;
  if (i < n) dst[i] = (_Float16)src[i];
}

// xf = spike(bn(x)/thr >= run[h]) over [8192 tokens][768 ch], h = c>>6
__global__ void k_prep_xf(const float* __restrict__ x,
                          const float* __restrict__ g, const float* __restrict__ bb,
                          const float* __restrict__ m, const float* __restrict__ v,
                          const float* __restrict__ thr, const float* __restrict__ run,
                          _Float16* __restrict__ xf, int total) {
  int i = blockIdx.x * blockDim.x + threadIdx.x;
  if (i >= total) return;
  int c = i % 768;
  int h = c >> 6;
  float sc = g[h] * rsqrtf(v[h] + EPSF);
  float xn = ((x[i] - m[h]) * sc + bb[h]) / thr[0];
  xf[i] = (_Float16)((xn >= run[h]) ? 1.0f : 0.0f);
}

// ---------------------------------------------------------------------------
// GEMM 1: qkv = xf[8192x768] @ qkv_w^T[768x2304], fused BN+spike epilogue.
// q stored [96][1024][64]; k,v stored TRANSPOSED: kT/vT [96][64][1024] (binary).
// grid (9, 512), block 128 (4 waves, each wave = 16 rows x 64 cols)
// ---------------------------------------------------------------------------
__global__ __launch_bounds__(128) void k_gemm_qkv(
    const _Float16* __restrict__ xf, const _Float16* __restrict__ w16,
    _Float16* __restrict__ qb, _Float16* __restrict__ kT, _Float16* __restrict__ vT,
    const float* __restrict__ nk_g, const float* __restrict__ nk_b,
    const float* __restrict__ nk_m, const float* __restrict__ nk_v,
    const float* __restrict__ ak_thr, const float* __restrict__ ak_run,
    const float* __restrict__ nv_g, const float* __restrict__ nv_b,
    const float* __restrict__ nv_m, const float* __restrict__ nv_v,
    const float* __restrict__ av_thr, const float* __restrict__ av_run) {
  const int lane = threadIdx.x & 31;
  const int wave = threadIdx.x >> 5;
  const int m0 = blockIdx.y * 16;
  const int j0 = blockIdx.x * 256 + wave * 64;

  v8f acc[4] = {};
  const _Float16* ap = xf + m0 * 768;
  for (int k0 = 0; k0 < 768; k0 += 32) {
    __builtin_prefetch(ap + k0 + 32, 0, 1);
    v16h a = ld_a16x32(ap + k0, 768, lane);
#pragma unroll
    for (int s = 0; s < 4; ++s) {
      v16h b = ld_b32x16(w16 + (j0 + s * 16) * 768 + k0, 768, lane);
      acc[s] = __builtin_amdgcn_wmma_f32_16x16x32_f16(false, a, false, b,
                                                      (short)0, acc[s], false, false);
    }
  }

  const int l16 = lane & 15, hl = lane >> 4;
#pragma unroll
  for (int s = 0; s < 4; ++s) {
    const int j = j0 + s * 16 + l16;
#pragma unroll
    for (int r = 0; r < 8; ++r) {
      const int t = m0 + r + hl * 8;         // token row in [0,8192)
      const int bi = t >> 10, tok = t & 1023;
      const float val = acc[s][r];
      if (j < 768) {                          // q: passthrough
        const int h = j >> 6, d = j & 63;
        qb[((bi * 12 + h) * 1024 + tok) * 64 + d] = (_Float16)val;
      } else if (j < 1536) {                  // k: bn + spike, store transposed
        const int jj = j - 768, h = jj >> 6, d = jj & 63;
        const float sc = nk_g[h] * rsqrtf(nk_v[h] + EPSF);
        const float xn = ((val - nk_m[h]) * sc + nk_b[h]) / ak_thr[0];
        kT[((bi * 12 + h) * 64 + d) * 1024 + tok] =
            (_Float16)((xn >= ak_run[h]) ? 1.0f : 0.0f);
      } else {                                // v: bn + spike, store transposed
        const int jj = j - 1536, h = jj >> 6, d = jj & 63;
        const float sc = nv_g[h] * rsqrtf(nv_v[h] + EPSF);
        const float xn = ((val - nv_m[h]) * sc + nv_b[h]) / av_thr[0];
        vT[((bi * 12 + h) * 64 + d) * 1024 + tok] =
            (_Float16)((xn >= av_run[h]) ? 1.0f : 0.0f);
      }
    }
  }
}

// ---------------------------------------------------------------------------
// GEMM 2: W = k^T @ v per (b,h); store W^T (f16, exact small integers).
// A = kT[bh] rows (contiguous tokens), Bt = vT[bh] rows (contiguous tokens).
// grid (96), block 128: wave w -> W rows [16w,16w+16) x 64 cols, K=1024 tokens.
// ---------------------------------------------------------------------------
__global__ __launch_bounds__(128) void k_gemm_ktv(
    const _Float16* __restrict__ kT, const _Float16* __restrict__ vT,
    _Float16* __restrict__ Wt) {
  const int bh = blockIdx.x;
  const int lane = threadIdx.x & 31;
  const int wave = threadIdx.x >> 5;
  const _Float16* kp = kT + bh * 64 * 1024;
  const _Float16* vp = vT + bh * 64 * 1024;
  const int m0 = wave * 16;

  v8f acc[4] = {};
  for (int t0 = 0; t0 < 1024; t0 += 32) {
    __builtin_prefetch(kp + m0 * 1024 + t0 + 32, 0, 1);
    v16h a = ld_a16x32(kp + m0 * 1024 + t0, 1024, lane);
#pragma unroll
    for (int s = 0; s < 4; ++s) {
      v16h b = ld_b32x16(vp + (s * 16) * 1024 + t0, 1024, lane);
      acc[s] = __builtin_amdgcn_wmma_f32_16x16x32_f16(false, a, false, b,
                                                      (short)0, acc[s], false, false);
    }
  }

  const int l16 = lane & 15, hl = lane >> 4;
#pragma unroll
  for (int s = 0; s < 4; ++s) {
#pragma unroll
    for (int r = 0; r < 8; ++r) {
      const int mm = m0 + r + hl * 8;   // dk index
      const int n = s * 16 + l16;       // dout index
      Wt[bh * 4096 + n * 64 + mm] = (_Float16)acc[s][r];  // W^T[n][m], exact
    }
  }
}

// ---------------------------------------------------------------------------
// GEMM 3: y = q[1024x64] @ W[64x64] per (b,h); epilogue bakes in the
// reshape(B,H,D,N) BN + spike + reshape(B,N,C), writing binary z[8192x768].
// grid (96, 16), block 128: wave -> 16 token rows x 64 cols, K=64.
// ---------------------------------------------------------------------------
__global__ __launch_bounds__(128) void k_gemm_qw(
    const _Float16* __restrict__ qb, const _Float16* __restrict__ Wt,
    _Float16* __restrict__ z,
    const float* __restrict__ no_g, const float* __restrict__ no_b,
    const float* __restrict__ no_m, const float* __restrict__ no_v,
    const float* __restrict__ ao_thr, const float* __restrict__ ao_run) {
  const int bh = blockIdx.x;
  const int lane = threadIdx.x & 31;
  const int wave = threadIdx.x >> 5;
  const int m0 = blockIdx.y * 64 + wave * 16;
  const _Float16* qp = qb + bh * 1024 * 64;
  const _Float16* wp = Wt + bh * 4096;

  v8f acc[4] = {};
#pragma unroll
  for (int k0 = 0; k0 < 64; k0 += 32) {
    v16h a = ld_a16x32(qp + m0 * 64 + k0, 64, lane);
#pragma unroll
    for (int s = 0; s < 4; ++s) {
      v16h b = ld_b32x16(wp + (s * 16) * 64 + k0, 64, lane);
      acc[s] = __builtin_amdgcn_wmma_f32_16x16x32_f16(false, a, false, b,
                                                      (short)0, acc[s], false, false);
    }
  }

  const int bi = bh / 12, h = bh % 12;
  const float sc = no_g[h] * rsqrtf(no_v[h] + EPSF);
  const float mu = no_m[h], bb = no_b[h];
  const float ithr = 1.0f / ao_thr[0], rth = ao_run[h];
  const int l16 = lane & 15, hl = lane >> 4;
#pragma unroll
  for (int s = 0; s < 4; ++s) {
#pragma unroll
    for (int r = 0; r < 8; ++r) {
      const int tok = m0 + r + hl * 8;
      const int d = s * 16 + l16;
      const int f = tok * 64 + d;       // flat offset within (b,h) block
      const int n2 = f & 1023;          // new token index after reshape
      const int d2 = f >> 10;           // new channel-within-head
      const float xn = ((acc[s][r] - mu) * sc + bb) * ithr;
      z[(bi * 1024 + n2) * 768 + h * 64 + d2] =
          (_Float16)((xn >= rth) ? 1.0f : 0.0f);
    }
  }
}

// ---------------------------------------------------------------------------
// GEMM 4: out = z[8192x768] @ proj_w^T[768x768] + proj_b, f32 output.
// grid (3, 512), block 128.
// ---------------------------------------------------------------------------
__global__ __launch_bounds__(128) void k_gemm_proj(
    const _Float16* __restrict__ z, const _Float16* __restrict__ p16,
    const float* __restrict__ pb, float* __restrict__ out) {
  const int lane = threadIdx.x & 31;
  const int wave = threadIdx.x >> 5;
  const int m0 = blockIdx.y * 16;
  const int j0 = blockIdx.x * 256 + wave * 64;

  v8f acc[4] = {};
  const _Float16* ap = z + m0 * 768;
  for (int k0 = 0; k0 < 768; k0 += 32) {
    __builtin_prefetch(ap + k0 + 32, 0, 1);
    v16h a = ld_a16x32(ap + k0, 768, lane);
#pragma unroll
    for (int s = 0; s < 4; ++s) {
      v16h b = ld_b32x16(p16 + (j0 + s * 16) * 768 + k0, 768, lane);
      acc[s] = __builtin_amdgcn_wmma_f32_16x16x32_f16(false, a, false, b,
                                                      (short)0, acc[s], false, false);
    }
  }

  const int l16 = lane & 15, hl = lane >> 4;
#pragma unroll
  for (int s = 0; s < 4; ++s) {
    const int j = j0 + s * 16 + l16;
    const float bias = pb[j];
#pragma unroll
    for (int r = 0; r < 8; ++r) {
      const int t = m0 + r + hl * 8;
      out[t * 768 + j] = acc[s][r] + bias;
    }
  }
}

// ---------------------------------------------------------------------------
// Host launcher
// ---------------------------------------------------------------------------
extern "C" void kernel_launch(void* const* d_in, const int* in_sizes, int n_in,
                              void* d_out, int out_size, void* d_ws, size_t ws_size,
                              hipStream_t stream) {
  const float* x      = (const float*)d_in[0];
  const float* qkv_w  = (const float*)d_in[1];
  const float* proj_w = (const float*)d_in[2];
  const float* proj_b = (const float*)d_in[3];
  const float* n_g  = (const float*)d_in[4];
  const float* n_b  = (const float*)d_in[5];
  const float* n_m  = (const float*)d_in[6];
  const float* n_v  = (const float*)d_in[7];
  const float* nk_g = (const float*)d_in[8];
  const float* nk_b = (const float*)d_in[9];
  const float* nk_m = (const float*)d_in[10];
  const float* nk_v = (const float*)d_in[11];
  const float* nv_g = (const float*)d_in[12];
  const float* nv_b = (const float*)d_in[13];
  const float* nv_m = (const float*)d_in[14];
  const float* nv_v = (const float*)d_in[15];
  const float* no_g = (const float*)d_in[16];
  const float* no_b = (const float*)d_in[17];
  const float* no_m = (const float*)d_in[18];
  const float* no_v = (const float*)d_in[19];
  const float* a_thr  = (const float*)d_in[20];
  const float* a_run  = (const float*)d_in[21];
  const float* ak_thr = (const float*)d_in[22];
  const float* ak_run = (const float*)d_in[23];
  const float* av_thr = (const float*)d_in[24];
  const float* av_run = (const float*)d_in[25];
  const float* ao_thr = (const float*)d_in[26];
  const float* ao_run = (const float*)d_in[27];

  // workspace layout (bytes); xf buffer is reused for z (xf fully consumed
  // by k_gemm_qkv before k_gemm_qw writes z)
  char* ws = (char*)d_ws;
  _Float16* xf  = (_Float16*)(ws + 0);          // [8192][768]  12.58 MB (also z)
  _Float16* w16 = (_Float16*)(ws + 12582912);   // [2304][768]   3.54 MB
  _Float16* p16 = (_Float16*)(ws + 16121856);   // [768][768]    1.18 MB
  _Float16* qb  = (_Float16*)(ws + 17301504);   // [96][1024][64]
  _Float16* kT  = (_Float16*)(ws + 29884416);   // [96][64][1024]
  _Float16* vT  = (_Float16*)(ws + 42467328);   // [96][64][1024]
  _Float16* Wt  = (_Float16*)(ws + 55050240);   // [96][64][64]
  if (ws_size < (size_t)55836672) return;

  k_cvt_f16<<<(1769472 + 255) / 256, 256, 0, stream>>>(qkv_w, w16, 1769472);
  k_cvt_f16<<<(589824 + 255) / 256, 256, 0, stream>>>(proj_w, p16, 589824);
  k_prep_xf<<<(6291456 + 255) / 256, 256, 0, stream>>>(
      x, n_g, n_b, n_m, n_v, a_thr, a_run, xf, 6291456);

  dim3 g1(9, 512);
  k_gemm_qkv<<<g1, 128, 0, stream>>>(xf, w16, qb, kT, vT,
                                     nk_g, nk_b, nk_m, nk_v, ak_thr, ak_run,
                                     nv_g, nv_b, nv_m, nv_v, av_thr, av_run);

  k_gemm_ktv<<<96, 128, 0, stream>>>(kT, vT, Wt);

  dim3 g3(96, 16);
  k_gemm_qw<<<g3, 128, 0, stream>>>(qb, Wt, xf /* z */,
                                    no_g, no_b, no_m, no_v, ao_thr, ao_run);

  dim3 g4(3, 512);
  k_gemm_proj<<<g4, 128, 0, stream>>>(xf /* z */, p16, proj_b, (float*)d_out);
}
